// SelfAttention_8126078124754
// MI455X (gfx1250) — matrix operand order repaired and checked by reading, not compiled, not run on hardware
//
#include <hip/hip_runtime.h>
#include <hip/hip_bf16.h>
#include <math.h>

// ---------------------------------------------------------------------------
// Self-attention (B=4, C=256, H=W=64) as fused flash-attention on CDNA5 WMMA.
//
//  Pass 0: transpose + f16-convert the 1x1-conv weights (Wt[c][o] = W[o][c]).
//  Pass 1: WMMA projection. 4 waves/block; per-k-chunk weight rows staged in
//          LDS via GLOBAL_LOAD_ASYNC_TO_LDS_B128, double-buffered.
//  Pass 2: flash attention. 4 waves/block, each wave owns 16 query rows.
//          K/V j-blocks double-buffered in LDS via async copies (ASYNCcnt),
//          all WMMA B-operands from LDS with conflict-padded pitches,
//          epilogue transposed through LDS for coalesced float4 x/out I/O.
//
// WMMA layout assumptions (CDNA5 ISA 7.12.2, wave32):
//   A (16x32 f16): lane = row M (lane&15); slots 0-7  -> K = 8*(lane>>4)+t,
//                                          slots 8-15 -> K = 16+8*(lane>>4)+t.
//   B (32x16 f16): lane = col N (lane&15); slots hold K in the same order as A.
//   C/D (16x16 f32): vgpr r, lane l: row = r + 8*(l>>4), col = l&15.
// ---------------------------------------------------------------------------

typedef __attribute__((ext_vector_type(16))) _Float16 v16h;
typedef __attribute__((ext_vector_type(8)))  _Float16 v8h;
typedef __attribute__((ext_vector_type(8)))  float    v8f;

#define B_SZ   4
#define C_DIM  256
#define D_QK   32
#define N_PIX  4096   // 64*64

static __device__ __forceinline__ v8f wmma_f16(v16h a, v16h b, v8f c) {
  return __builtin_amdgcn_wmma_f32_16x16x32_f16(false, a, false, b,
                                                (short)0, c, false, false);
}

// 16B global -> LDS async copy (per active lane), tracked by ASYNCcnt.
static __device__ __forceinline__ void async_copy16(unsigned lds_off,
                                                    const void* gaddr) {
  asm volatile("global_load_async_to_lds_b128 %0, %1, off"
               :: "v"(lds_off), "v"((unsigned long long)(uintptr_t)gaddr)
               : "memory");
}
static __device__ __forceinline__ void wait_async0() {
  asm volatile("s_wait_asynccnt 0x0" ::: "memory");
}
static __device__ __forceinline__ void wait_ds0() {
  asm volatile("s_wait_dscnt 0x0" ::: "memory");
}
// Gather a B column from LDS: p = column base (k=0), elements k = 8h+t / 16+8h+t.
static __device__ __forceinline__ v16h ld_tile(const _Float16* p, int pitch, int h) {
  v16h v;
#pragma unroll
  for (int t = 0; t < 8; ++t) {
    v[t]     = p[(8 * h + t) * pitch];
    v[8 + t] = p[(16 + 8 * h + t) * pitch];
  }
  return v;
}

// ---------------------------------------------------------------------------
// Pass 0: Wt[c][o] = (f16) W[o][c]
// ---------------------------------------------------------------------------
__global__ void wt_cvt_kernel(const float* __restrict__ W,
                              _Float16* __restrict__ Wt, int O, int Cc) {
  int i = blockIdx.x * blockDim.x + threadIdx.x;
  if (i >= O * Cc) return;
  int o = i / Cc, c = i - o * Cc;
  Wt[(size_t)c * O + o] = (_Float16)W[(size_t)o * Cc + c];
}

// ---------------------------------------------------------------------------
// Pass 1: projections. 4 waves/block, each wave = 16 pixels, 320 out chans.
// Weight rows for the current 32-channel k-chunk live in LDS:
//   row layout (per k-row, 328 halves = 656B pitch, 8-half pad):
//     halves [0,32) = WtQ row, [32,64) = WtK row, [64,320) = WtV row.
// ---------------------------------------------------------------------------
#define PW_PITCH_H 328                 // halves per weight row (320+8 pad)
#define PW_PITCH_B 656
#define PLDS_W(p)  ((p) * 20992)       // 32 rows * 656B
#define PLDS_TOTAL 41984

__global__ __launch_bounds__(128)
void proj_qkv_kernel(const float* __restrict__ x,      // [B][C][N]
                     const _Float16* __restrict__ WtQ, // [C][32]
                     const _Float16* __restrict__ WtK, // [C][32]
                     const _Float16* __restrict__ WtV, // [C][256]
                     const float* __restrict__ bq,
                     const float* __restrict__ bk,
                     const float* __restrict__ bv,
                     _Float16* __restrict__ Qb,        // [B][N][32]
                     _Float16* __restrict__ Kt,        // [B][32][N]
                     _Float16* __restrict__ Vb)        // [B][N][256]
{
  __shared__ __align__(16) char smem[PLDS_TOTAL];

  const int tid  = threadIdx.x;
  const int lane = tid & 31;
  const int wave = tid >> 5;
  const int b    = blockIdx.y;
  const int n0   = blockIdx.x * 64 + wave * 16;
  const int ml   = lane & 15;
  const int h    = lane >> 4;
  const unsigned smem_base = (unsigned)(uintptr_t)(void*)smem;

  // Stage k-chunk [kk, kk+32) weight rows into LDS buffer p.
  // 32 rows x 40 chunks(16B) = 1280 chunks; 10 per thread.
  auto issue_w = [&](int kk, int p) {
#pragma unroll
    for (int i = 0; i < 10; ++i) {
      int c = tid + 128 * i;
      int row = c / 40, part = c % 40;
      const char* src;
      if (part < 4)      src = (const char*)WtQ + (size_t)(kk + row) * 64  + part * 16;
      else if (part < 8) src = (const char*)WtK + (size_t)(kk + row) * 64  + (part - 4) * 16;
      else               src = (const char*)WtV + (size_t)(kk + row) * 512 + (part - 8) * 16;
      async_copy16(smem_base + PLDS_W(p) + row * PW_PITCH_B + part * 16, src);
    }
  };

  v8f accQ[2], accK[2], accV[16];
#pragma unroll
  for (int t = 0; t < 2; ++t)  { accQ[t] = {}; accK[t] = {}; }
#pragma unroll
  for (int t = 0; t < 16; ++t) accV[t] = {};

  const float* xb = x + (size_t)b * C_DIM * N_PIX;

  issue_w(0, 0);
  wait_async0();
  __syncthreads();

  int p = 0;
  for (int kk = 0; kk < C_DIM; kk += 32) {
    if (kk + 32 < C_DIM) issue_w(kk + 32, p ^ 1);

    // A = x^T tile [16 pixels x 32 channels], strided gather from x[b][c][n]
    v16h a;
#pragma unroll
    for (int t = 0; t < 8; ++t) {
      a[t]     = (_Float16)xb[(size_t)(kk + 8 * h + t)      * N_PIX + n0 + ml];
      a[8 + t] = (_Float16)xb[(size_t)(kk + 16 + 8 * h + t) * N_PIX + n0 + ml];
    }

    const _Float16* wl = (const _Float16*)(smem + PLDS_W(p)) + ml;
#pragma unroll
    for (int tq = 0; tq < 2; ++tq)
      accQ[tq] = wmma_f16(a, ld_tile(wl + 16 * tq, PW_PITCH_H, h), accQ[tq]);
#pragma unroll
    for (int tk = 0; tk < 2; ++tk)
      accK[tk] = wmma_f16(a, ld_tile(wl + 32 + 16 * tk, PW_PITCH_H, h), accK[tk]);
#pragma unroll
    for (int tv = 0; tv < 16; ++tv)
      accV[tv] = wmma_f16(a, ld_tile(wl + 64 + 16 * tv, PW_PITCH_H, h), accV[tv]);

    wait_async0();
    __syncthreads();
    p ^= 1;
  }

  // Store with bias.  D-tile: row = r + 8h (pixel), col = ml (out channel).
#pragma unroll
  for (int tq = 0; tq < 2; ++tq)
#pragma unroll
    for (int r = 0; r < 8; ++r) {
      int row = r + 8 * h, col = ml + 16 * tq;
      Qb[((size_t)b * N_PIX + n0 + row) * D_QK + col] =
          (_Float16)(accQ[tq][r] + bq[col]);
      Kt[((size_t)b * D_QK + col) * N_PIX + n0 + row] =
          (_Float16)(accK[tq][r] + bk[col]);
    }
#pragma unroll
  for (int tv = 0; tv < 16; ++tv)
#pragma unroll
    for (int r = 0; r < 8; ++r) {
      int row = r + 8 * h, col = ml + 16 * tv;
      Vb[((size_t)b * N_PIX + n0 + row) * C_DIM + col] =
          (_Float16)(accV[tv][r] + bv[col]);
    }
}

// ---------------------------------------------------------------------------
// Pass 2: flash attention.  4 waves/block; wave w owns query rows
// [blockIdx.x*64 + 16w, +16).  Conflict-padded LDS pitches (odd 16B-group
// shift per lane => 2-way max on 16B accesses).
// LDS map (bytes):
//   [     0,  5120)  K bufs: 2 x 32 rows x 80B  (64B data + 16B pad)
//   [  5120, 38912)  V bufs: 2 x 32 rows x 528B (512B data + 16B pad)
//   [ 38912, 44032)  P: 4 waves x 16 rows x 80B
//   [ 44032, 52736)  O epilogue: 4 waves x 16 x 33 f32 (2112B, padded 2176)
// ---------------------------------------------------------------------------
#define KP_H 40                        // K/P row pitch in halves (80B)
#define VP_H 264                       // V row pitch in halves (528B)
#define LDS_K(p)   ((p) * 2560)
#define LDS_V(p)   (5120 + (p) * 16896)
#define LDS_P(w)   (38912 + (w) * 1280)
#define LDS_O(w)   (44032 + (w) * 2176)
#define LDS_TOTAL  52736

__global__ __launch_bounds__(128)
void attn_kernel(const float* __restrict__ x,       // [B][C][N]
                 const _Float16* __restrict__ Qb,   // [B][N][32]
                 const _Float16* __restrict__ Kt,   // [B][32][N]
                 const _Float16* __restrict__ Vb,   // [B][N][256]
                 const float* __restrict__ gamma,
                 float* __restrict__ out)           // [B][C][N]
{
  __shared__ __align__(16) char smem[LDS_TOTAL];

  const int tid  = threadIdx.x;
  const int lane = tid & 31;
  const int wave = tid >> 5;
  const int b    = blockIdx.y;
  const int n0   = blockIdx.x * 64 + wave * 16;
  const int ml   = lane & 15;
  const int h    = lane >> 4;
  const unsigned smem_base = (unsigned)(uintptr_t)(void*)smem;

  const _Float16* ktb = Kt + (size_t)b * D_QK * N_PIX;
  const _Float16* vbb = Vb + (size_t)b * N_PIX * C_DIM;

  // Stage j-block j0 into buffer p (whole workgroup).
  auto issue_async = [&](int j0, int p) {
    {  // K block: 32 rows x 64B = 128 x 16B chunks, one per thread
      int d = tid >> 2, sub = tid & 3;
      async_copy16(smem_base + LDS_K(p) + d * 80 + sub * 16,
                   (const char*)(ktb + (size_t)d * N_PIX + j0) + sub * 16);
    }
#pragma unroll
    for (int r = 0; r < 8; ++r) {  // V block: 32 rows x 512B = 1024 chunks
      int c = tid + 128 * r;
      int j = c >> 5, sub = c & 31;
      async_copy16(smem_base + LDS_V(p) + j * 528 + sub * 16,
                   (const char*)(vbb + (size_t)(j0 + j) * C_DIM) + sub * 16);
    }
  };

  // Q A-tile: lane ml holds query row n0+ml (two 16B chunks per A layout).
  v16h aQ;
  const _Float16* qrow = Qb + ((size_t)b * N_PIX + n0 + ml) * D_QK;
#pragma unroll
  for (int t = 0; t < 8; ++t) {
    aQ[t]     = qrow[8 * h + t];
    aQ[8 + t] = qrow[16 + 8 * h + t];
  }

  float mrun[8], lrun[8];
  v8f accO[16];
#pragma unroll
  for (int t = 0; t < 8; ++t) { mrun[t] = -__builtin_inff(); lrun[t] = 0.0f; }
#pragma unroll
  for (int ct = 0; ct < 16; ++ct) accO[ct] = {};

  _Float16* ldsP = (_Float16*)(smem + LDS_P(wave));
  const v8f zero = {};

  issue_async(0, 0);
  wait_async0();
  __syncthreads();

  int p = 0;
  for (int j0 = 0; j0 < N_PIX; j0 += 32) {
    if (j0 + 32 < N_PIX) issue_async(j0 + 32, p ^ 1);   // prefetch next block

    const _Float16* ldsK = (const _Float16*)(smem + LDS_K(p));
    const _Float16* ldsV = (const _Float16*)(smem + LDS_V(p));

    // S tiles: A=Q [16x32], B=K [32 d x 16 j]; B lane = j-column, elements = d (from LDS).
    const _Float16* krow = ldsK + ml;
    v8f s0 = wmma_f16(aQ, ld_tile(krow, KP_H, h), zero);
    v8f s1 = wmma_f16(aQ, ld_tile(krow + 16, KP_H, h), zero);

    // Online softmax. D-tile row r+8h lives inside the lane's 16-lane half,
    // so xor-butterflies over masks 1,2,4,8 reduce exactly one row.
    float mb[8];
#pragma unroll
    for (int t = 0; t < 8; ++t) {
      mb[t] = fmaxf(s0[t], s1[t]);
#pragma unroll
      for (int msk = 1; msk <= 8; msk <<= 1)
        mb[t] = fmaxf(mb[t], __shfl_xor(mb[t], msk, 32));
    }
    float scal[8], rs[8];
#pragma unroll
    for (int t = 0; t < 8; ++t) {
      float mnew = fmaxf(mrun[t], mb[t]);
      scal[t] = __expf(mrun[t] - mnew);
      float p0 = __expf(s0[t] - mnew);
      float p1 = __expf(s1[t] - mnew);
      rs[t] = p0 + p1;
      ldsP[(t + 8 * h) * KP_H + ml]      = (_Float16)p0;  // P row-major (f16)
      ldsP[(t + 8 * h) * KP_H + 16 + ml] = (_Float16)p1;
      mrun[t] = mnew;
    }
#pragma unroll
    for (int t = 0; t < 8; ++t) {
      float r_ = rs[t];
#pragma unroll
      for (int msk = 1; msk <= 8; msk <<= 1) r_ += __shfl_xor(r_, msk, 32);
      lrun[t] = lrun[t] * scal[t] + r_;
    }
#pragma unroll
    for (int ct = 0; ct < 16; ++ct)
#pragma unroll
      for (int t = 0; t < 8; ++t) accO[ct][t] *= scal[t];

    // Intra-wave LDS transpose of P (lane = col -> lane = row). DS ops are
    // program-ordered within the wave; fence keeps the compiler honest.
    wait_ds0();
    v16h aP;
    {
      const _Float16* prow = ldsP + ml * KP_H;
      union { v16h v; v8h hh[2]; } u;
      u.hh[0] = *(const v8h*)(prow + 8 * h);
      u.hh[1] = *(const v8h*)(prow + 16 + 8 * h);
      aP = u.v;
    }
    wait_ds0();

    // O += P * V.  B lane = channel column, elements = j-local, from padded LDS.
    const _Float16* vrow = ldsV + ml;
#pragma unroll
    for (int ct = 0; ct < 16; ++ct)
      accO[ct] = wmma_f16(aP, ld_tile(vrow + 16 * ct, VP_H, h), accO[ct]);

    wait_async0();       // next block fully staged (this wave's share)
    __syncthreads();     // all waves done computing on p and staging p^1
    p ^= 1;
  }

  // Epilogue: out[b][c][n] = x[b][c][n] + gamma * O/l, transposed through
  // LDS so each lane owns one channel and does coalesced float4 I/O.
  float* ldsO = (float*)(smem + LDS_O(wave));
  const float g = gamma[0];
  float inv[8];
#pragma unroll
  for (int t = 0; t < 8; ++t) inv[t] = 1.0f / lrun[t];

  for (int gi = 0; gi < 8; ++gi) {          // 32 channels per pass
#pragma unroll
    for (int tt = 0; tt < 2; ++tt)
#pragma unroll
      for (int t = 0; t < 8; ++t)
        ldsO[(t + 8 * h) * 33 + 16 * tt + ml] = accO[2 * gi + tt][t] * inv[t];
    wait_ds0();
    int c = 32 * gi + lane;
    const float* xr = x   + ((size_t)b * C_DIM + c) * N_PIX + n0;
    float*      orw = out + ((size_t)b * C_DIM + c) * N_PIX + n0;
#pragma unroll
    for (int q = 0; q < 4; ++q) {
      float4 xv = *(const float4*)(xr + 4 * q);
      float4 ov;
      ov.x = xv.x + g * ldsO[(4 * q + 0) * 33 + lane];
      ov.y = xv.y + g * ldsO[(4 * q + 1) * 33 + lane];
      ov.z = xv.z + g * ldsO[(4 * q + 2) * 33 + lane];
      ov.w = xv.w + g * ldsO[(4 * q + 3) * 33 + lane];
      *(float4*)(orw + 4 * q) = ov;
    }
    wait_ds0();
  }
}

// ---------------------------------------------------------------------------
extern "C" void kernel_launch(void* const* d_in, const int* in_sizes, int n_in,
                              void* d_out, int out_size, void* d_ws, size_t ws_size,
                              hipStream_t stream) {
  (void)in_sizes; (void)n_in; (void)out_size; (void)ws_size;
  const float* x     = (const float*)d_in[0];
  const float* Wq    = (const float*)d_in[1];
  const float* bq    = (const float*)d_in[2];
  const float* Wk    = (const float*)d_in[3];
  const float* bk    = (const float*)d_in[4];
  const float* Wv    = (const float*)d_in[5];
  const float* bv    = (const float*)d_in[6];
  const float* gamma = (const float*)d_in[7];
  float* out = (float*)d_out;

  char* ws = (char*)d_ws;
  size_t off = 0;
  auto carve = [&](size_t bytes) {
    char* p = ws + off;
    off = (off + bytes + 255) & ~(size_t)255;
    return p;
  };
  _Float16* WtQ = (_Float16*)carve((size_t)C_DIM * D_QK  * 2);
  _Float16* WtK = (_Float16*)carve((size_t)C_DIM * D_QK  * 2);
  _Float16* WtV = (_Float16*)carve((size_t)C_DIM * C_DIM * 2);
  _Float16* Qb  = (_Float16*)carve((size_t)B_SZ * N_PIX * D_QK  * 2);
  _Float16* Kt  = (_Float16*)carve((size_t)B_SZ * N_PIX * D_QK  * 2);
  _Float16* Vb  = (_Float16*)carve((size_t)B_SZ * N_PIX * C_DIM * 2);

  // Pass 0: weight transpose + f16 convert
  wt_cvt_kernel<<<(D_QK * C_DIM + 255) / 256, 256, 0, stream>>>(Wq, WtQ, D_QK, C_DIM);
  wt_cvt_kernel<<<(D_QK * C_DIM + 255) / 256, 256, 0, stream>>>(Wk, WtK, D_QK, C_DIM);
  wt_cvt_kernel<<<(C_DIM * C_DIM + 255) / 256, 256, 0, stream>>>(Wv, WtV, C_DIM, C_DIM);

  // Pass 1: QKV projections (WMMA, async LDS weight staging)
  proj_qkv_kernel<<<dim3(N_PIX / 64, B_SZ), 128, 0, stream>>>(
      x, WtQ, WtK, WtV, bq, bk, bv, Qb, Kt, Vb);
  // Pass 2: flash attention + residual (WMMA, async LDS double-buffering)
  attn_kernel<<<dim3(N_PIX / 64, B_SZ), 128, 0, stream>>>(
      x, Qb, Kt, Vb, gamma, out);
}
